// BitVocoder_68513318306183
// MI455X (gfx1250) — matrix operand order repaired and checked by reading, not compiled, not run on hardware
//
#include <hip/hip_runtime.h>
#include <hip/hip_bf16.h>
#include <math.h>
#include <stdint.h>

// ---------------- model constants ----------------
#define BATCH   8
#define TSEQ    2048
#define MROWS   (BATCH * TSEQ)          // 16384
#define DIMC    256
#define ICDIM   1024
#define FBINS   513
#define FBPAD   576                     // 513 -> mult of 64
#define NH      4
#define PROJN   2052
#define PROJPAD 2112                    // 2052 -> mult of 64
#define NFFT    1024
#define HOP     320
#define KISTFT  1056                    // 1026 -> mult of 32
#define LFULL   (NFFT + (TSEQ - 1) * HOP)   // 656064
#define LOUT    (LFULL - NFFT)              // 655040
#define PADT    2064                    // TSEQ + 16
#define TOFF    8

typedef __attribute__((ext_vector_type(16))) _Float16 v16h;
typedef __attribute__((ext_vector_type(8)))  _Float16 v8h;
typedef __attribute__((ext_vector_type(8)))  float    v8f;

// low 32 bits of a generic pointer into the LDS aperture == LDS byte offset
__device__ __forceinline__ unsigned lds_off32(const void* p)
{
    return (unsigned)(unsigned long long)(uintptr_t)p;
}

// =====================================================================
// Generic tapped WMMA GEMM:
//   Y[m, n] = bias[n] + sum_tap sum_k A[arow(m,tap)*K + k] * W[tap][n][k]
//   arow(m,tap) = (m>>11)*tStride + tOff + (m&2047) + (tap - ntaps/2)*dil
// A: f16 (zero-padded rows so arow always valid), W: f16 [ntaps][Npad][K],
// Y: f32 [M][Npad].  Block = 256 threads = 8 waves (2 M x 4 N), each wave
// computes a 64x16 strip as 4 stacked 16x16 WMMA tiles.
// Tiles are staged with the CDNA5 async data mover
// (global_load_async_to_lds_b128 + s_wait_asynccnt, tracked by ASYNCcnt).
// =====================================================================
__global__ __launch_bounds__(256) void wmma_gemm_taps(
    const _Float16* __restrict__ A, const _Float16* __restrict__ W,
    const float* __restrict__ bias, float* __restrict__ Y,
    int Npad, int K, int tStride, int tOff, int ntaps, int dil)
{
    __shared__ _Float16 lsA[128 * 32];   // 8 KB
    __shared__ _Float16 lsB[64 * 32];    // 4 KB

    const int tid    = threadIdx.x;
    const int lane   = tid & 31;
    const int wave   = tid >> 5;
    const int wm     = wave & 1;         // 2 waves along M
    const int wn     = wave >> 1;        // 4 waves along N
    const int m0     = blockIdx.x * 128;
    const int n0     = blockIdx.y * 64;
    const int lh     = lane >> 4;        // 0/1 half-wave
    const int l15    = lane & 15;

    v8f acc[4] = {};

    // staging indices (constant across k-loop)
    const int rA = tid >> 1, cA = tid & 1;         // 128 rows x 2 chunks
    const int mA = m0 + rA;
    const int bA = mA >> 11, tA = mA & (TSEQ - 1);
    const int rB = (tid & 127) >> 1, cB = tid & 1; // 64 rows x 2 chunks

    // per-lane LDS destinations (byte addresses) for the async loads
    const unsigned ldsA = lds_off32(&lsA[rA * 32 + cA * 16]);
    const unsigned ldsB = lds_off32(&lsB[rB * 32 + cB * 16]);
    const unsigned long long baseA = (unsigned long long)(uintptr_t)A;
    const unsigned long long baseW = (unsigned long long)(uintptr_t)W;

    for (int tap = 0; tap < ntaps; ++tap) {
        const int roff = (tap - (ntaps >> 1)) * dil;
        const long arow = (long)bA * tStride + tOff + tA + roff;
        // per-lane byte offsets into A / W for this tap (k0 added per step)
        const unsigned offA0 = (unsigned)((arow * K + cA * 16) * 2);
        const unsigned offB0 =
            (unsigned)((((long)tap * Npad + n0 + rB) * K + cB * 16) * 2);
        __builtin_prefetch(W + ((long)tap * Npad + n0 + rB) * K, 0, 3);

        for (int k0 = 0; k0 < K; k0 += 32) {
            const unsigned kb = (unsigned)(k0 * 2);
            // ---- async stage A tile: 256 lanes x 16B = 4 KB ----
            asm volatile("global_load_async_to_lds_b128 %0, %1, %2"
                         :: "v"(ldsA), "v"(offA0 + kb), "s"(baseA) : "memory");
            // ---- async stage B tile: 128 lanes x 16B = 2 KB ----
            if (tid < 128) {
                asm volatile("global_load_async_to_lds_b128 %0, %1, %2"
                             :: "v"(ldsB), "v"(offB0 + kb), "s"(baseW) : "memory");
            }
            asm volatile("s_wait_asynccnt 0" ::: "memory");
            __syncthreads();

            // ---- B fragment: N=l15; lane<16 -> K=0..7,16..23; lane>=16 -> K=8..15,24..31
            const int bn = wn * 16 + l15;
            v8h blo = *(const v8h*)&lsB[bn * 32 + lh * 8];
            v8h bhi = *(const v8h*)&lsB[bn * 32 + 16 + lh * 8];
            v16h bf = __builtin_shufflevector(blo, bhi,
                0, 1, 2, 3, 4, 5, 6, 7, 8, 9, 10, 11, 12, 13, 14, 15);

            // ---- 4 stacked A tiles along M ----
            #pragma unroll
            for (int tm = 0; tm < 4; ++tm) {
                const int am = wm * 64 + tm * 16 + l15;
                v8h lo = *(const v8h*)&lsA[am * 32 + lh * 8];
                v8h hi = *(const v8h*)&lsA[am * 32 + 16 + lh * 8];
                v16h af = __builtin_shufflevector(lo, hi,
                    0, 1, 2, 3, 4, 5, 6, 7, 8, 9, 10, 11, 12, 13, 14, 15);
                acc[tm] = __builtin_amdgcn_wmma_f32_16x16x32_f16(
                    false, af, false, bf, (short)0, acc[tm], false, false);
            }
            __syncthreads();
        }
    }

    const int col = n0 + wn * 16 + l15;
    const float bc = bias ? bias[col] : 0.f;
    #pragma unroll
    for (int tm = 0; tm < 4; ++tm) {
        const int row0 = m0 + wm * 64 + tm * 16 + lh * 8;
        #pragma unroll
        for (int v = 0; v < 8; ++v)
            Y[(long)(row0 + v) * Npad + col] = acc[tm][v] + bc;
    }
}

// =====================================================================
// ternarize + repack (O,I,Kt) f32 -> (Kt,Opad,I) f16, zero for o>=O
// =====================================================================
__global__ __launch_bounds__(256) void k_ternarize_repack(
    const float* __restrict__ w, _Float16* __restrict__ dst,
    int O, int I, int Kt, int Opad)
{
    __shared__ float red[256];
    __shared__ float s_scale;
    const int n = O * I * Kt;
    float s = 0.f;
    for (int i = threadIdx.x; i < n; i += 256) s += fabsf(w[i]);
    red[threadIdx.x] = s;
    __syncthreads();
    for (int off = 128; off > 0; off >>= 1) {
        if (threadIdx.x < off) red[threadIdx.x] += red[threadIdx.x + off];
        __syncthreads();
    }
    if (threadIdx.x == 0) s_scale = red[0] / (float)n + 1e-5f;
    __syncthreads();
    const float sc = s_scale;
    const int total = Kt * Opad * I;
    for (int idx = threadIdx.x; idx < total; idx += 256) {
        const int k = idx / (Opad * I);
        const int r = idx - k * (Opad * I);
        const int o = r / I;
        const int i = r - o * I;
        float v = 0.f;
        if (o < O) {
            const float x = w[(o * I + i) * Kt + k];
            float q = rintf(x / sc);
            q = fminf(1.f, fmaxf(-1.f, q));
            v = q * sc;
        }
        dst[idx] = (_Float16)v;
    }
}

__global__ void k_padbias(const float* __restrict__ s, float* __restrict__ d,
                          int N, int Npad)
{
    int i = blockIdx.x * blockDim.x + threadIdx.x;
    if (i < Npad) d[i] = (i < N) ? s[i] : 0.f;
}

// zero padding rows of a [BATCH][PADT][DIMC] f16 buffer
__global__ void k_zero_pads(_Float16* __restrict__ p)
{
    int idx = blockIdx.x * blockDim.x + threadIdx.x;   // 8*16*256
    if (idx >= BATCH * 16 * DIMC) return;
    const int c = idx & 255;
    const int r = idx >> 8;
    const int b = r >> 4;
    const int pr = r & 15;
    const int tt = (pr < TOFF) ? pr : (PADT - 16 + pr);
    p[((long)b * PADT + tt) * DIMC + c] = (_Float16)0.f;
}

// f32 [M][256] -> f16 padded interior
__global__ void k_cast_pad(const float* __restrict__ s, _Float16* __restrict__ d)
{
    long idx = (long)blockIdx.x * blockDim.x + threadIdx.x;
    if (idx >= (long)MROWS * DIMC) return;
    const int c = idx & 255;
    const long m = idx >> 8;
    const int b = (int)(m >> 11), t = (int)(m & (TSEQ - 1));
    d[((long)b * PADT + TOFF + t) * DIMC + c] = (_Float16)s[idx];
}

// rmsnorm (C=256) -> f16 padded interior; one wave32 per row
__global__ __launch_bounds__(256) void k_rmsnorm_pad(
    const float* __restrict__ x, const float* __restrict__ wgt,
    _Float16* __restrict__ d)
{
    const int wave = threadIdx.x >> 5, lane = threadIdx.x & 31;
    const long m = (long)blockIdx.x * 8 + wave;
    const float* row = x + m * DIMC;
    float v[8]; float ss = 0.f;
    #pragma unroll
    for (int i = 0; i < 8; ++i) { v[i] = row[lane + i * 32]; ss += v[i] * v[i]; }
    #pragma unroll
    for (int o = 16; o > 0; o >>= 1) ss += __shfl_xor(ss, o, 32);
    const float r = rsqrtf(ss * (1.f / 256.f) + 1e-6f);
    const int b = (int)(m >> 11), t = (int)(m & (TSEQ - 1));
    _Float16* out = d + ((long)b * PADT + TOFF + t) * DIMC;
    #pragma unroll
    for (int i = 0; i < 8; ++i)
        out[lane + i * 32] = (_Float16)(v[i] * r * wgt[lane + i * 32]);
}

// depthwise conv K=7, dilation dil, zero-pad in t
__global__ void k_dwconv(const float* __restrict__ x, const float* __restrict__ w,
                         const float* __restrict__ bias, float* __restrict__ y, int dil)
{
    long idx = (long)blockIdx.x * blockDim.x + threadIdx.x;
    if (idx >= (long)MROWS * DIMC) return;
    const int c = idx & 255;
    const long m = idx >> 8;
    const int b = (int)(m >> 11), t = (int)(m & (TSEQ - 1));
    float acc = bias[c];
    #pragma unroll
    for (int j = 0; j < 7; ++j) {
        const int tt = t + (j - 3) * dil;
        if (tt >= 0 && tt < TSEQ)
            acc += w[c * 7 + j] * x[(((long)b << 11) + tt) * DIMC + c];
    }
    y[idx] = acc;
}

// snake_beta in-place f32; N power-of-two channel mask
__global__ void k_snake(float* __restrict__ x, const float* __restrict__ la,
                        const float* __restrict__ lb, int mask, long total)
{
    long idx = (long)blockIdx.x * blockDim.x + threadIdx.x;
    if (idx >= total) return;
    const int c = (int)(idx & mask);
    const float a = __expf(la[c]), b = __expf(lb[c]);
    const float s = __sinf(a * x[idx]);
    x[idx] = x[idx] + (1.f / (b + 1e-9f)) * s * s;
}

// snake_beta f32 -> f16
__global__ void k_snake_h(const float* __restrict__ x, const float* __restrict__ la,
                          const float* __restrict__ lb, _Float16* __restrict__ y,
                          int mask, long total)
{
    long idx = (long)blockIdx.x * blockDim.x + threadIdx.x;
    if (idx >= total) return;
    const int c = (int)(idx & mask);
    const float a = __expf(la[c]), b = __expf(lb[c]);
    const float s = __sinf(a * x[idx]);
    y[idx] = (_Float16)(x[idx] + (1.f / (b + 1e-9f)) * s * s);
}

__global__ void k_axpy_gamma(float* __restrict__ h, const float* __restrict__ y,
                             const float* __restrict__ g, long total)
{
    long idx = (long)blockIdx.x * blockDim.x + threadIdx.x;
    if (idx >= total) return;
    h[idx] += g[idx & 255] * y[idx];
}

__global__ void k_add(float* __restrict__ a, const float* __restrict__ b, long total)
{
    long idx = (long)blockIdx.x * blockDim.x + threadIdx.x;
    if (idx >= total) return;
    a[idx] += b[idx];
}

__global__ void k_expclip(float* __restrict__ m, long total)
{
    long idx = (long)blockIdx.x * blockDim.x + threadIdx.x;
    if (idx >= total) return;
    m[idx] = __expf(fminf(10.f, fmaxf(-10.f, m[idx])));
}

// combine heads (softmax over 4) + tanh*pi*cumsum_w
__global__ void k_headcomb(const float* __restrict__ proj, const float* __restrict__ hw,
                           const float* __restrict__ cw, float* __restrict__ g)
{
    long idx = (long)blockIdx.x * blockDim.x + threadIdx.x;
    if (idx >= (long)MROWS * FBINS) return;
    const long m = idx / FBINS;
    const int f = (int)(idx - m * FBINS);
    const float h0 = hw[0], h1 = hw[1], h2 = hw[2], h3 = hw[3];
    const float mx = fmaxf(fmaxf(h0, h1), fmaxf(h2, h3));
    const float e0 = __expf(h0 - mx), e1 = __expf(h1 - mx),
                e2 = __expf(h2 - mx), e3 = __expf(h3 - mx);
    const float inv = 1.f / (e0 + e1 + e2 + e3);
    const float* p = proj + m * PROJPAD;
    const float s = (e0 * p[f] + e1 * p[FBINS + f] + e2 * p[2 * FBINS + f] +
                     e3 * p[3 * FBINS + f]) * inv;
    g[idx] = tanhf(s) * 3.14159265358979f * cw[f];
}

// cumsum over t per (b,f), add phase_bias
__global__ void k_cumsum(const float* __restrict__ g, const float* __restrict__ pb,
                         float* __restrict__ ph)
{
    int idx = blockIdx.x * blockDim.x + threadIdx.x;
    if (idx >= BATCH * FBINS) return;
    const int b = idx / FBINS, f = idx - (idx / FBINS) * FBINS;
    float acc = 0.f;
    const float bias = pb[f];
    for (int t = 0; t < TSEQ; ++t) {
        const long m = ((long)b << 11) + t;
        acc += g[m * FBINS + f];
        ph[m * FBINS + f] = acc + bias;
    }
}

// 5-tap smoothing along t (zero padded)
__global__ void k_smooth(const float* __restrict__ ph, const float* __restrict__ kk,
                         float* __restrict__ ps)
{
    long idx = (long)blockIdx.x * blockDim.x + threadIdx.x;
    if (idx >= (long)MROWS * FBINS) return;
    const long m = idx / FBINS;
    const int f = (int)(idx - m * FBINS);
    const int b = (int)(m >> 11), t = (int)(m & (TSEQ - 1));
    float acc = 0.f;
    #pragma unroll
    for (int j = 0; j < 5; ++j) {
        const int tt = t + j - 2;
        if (tt >= 0 && tt < TSEQ)
            acc += kk[j] * ph[(((long)b << 11) + tt) * FBINS + f];
    }
    ps[idx] = acc;
}

// build [re | im | 0pad] f16 rows for the iSTFT GEMM
__global__ void k_spec(const float* __restrict__ mag, const float* __restrict__ ps,
                       _Float16* __restrict__ sp)
{
    long idx = (long)blockIdx.x * blockDim.x + threadIdx.x;
    if (idx >= (long)MROWS * KISTFT) return;
    const long m = idx / KISTFT;
    const int j = (int)(idx - m * KISTFT);
    float v = 0.f;
    if (j < 2 * FBINS) {
        const int f = (j < FBINS) ? j : (j - FBINS);
        const float mg = mag[m * FBPAD + f];
        const float p  = ps[m * FBINS + f];
        v = (j < FBINS) ? mg * __cosf(p) : mg * __sinf(p);
    }
    sp[idx] = (_Float16)v;
}

// iSTFT basis [NFFT rows n][KISTFT cols], window folded in
__global__ void k_basis(_Float16* __restrict__ wb)
{
    long idx = (long)blockIdx.x * blockDim.x + threadIdx.x;
    if (idx >= (long)NFFT * KISTFT) return;
    const int n = (int)(idx / KISTFT);
    const int j = (int)(idx - (long)n * KISTFT);
    const float win = 0.5f - 0.5f * __cosf(6.28318530717958647f * n * (1.f / NFFT));
    float v = 0.f;
    if (j < 2 * FBINS) {
        const int f = (j < FBINS) ? j : (j - FBINS);
        const float ck = (f == 0 || f == NFFT / 2) ? 1.f : 2.f;
        const int r = (f * n) & (NFFT - 1);
        const float ang = 6.28318530717958647f * (float)r * (1.f / NFFT);
        v = ((j < FBINS) ? ck * __cosf(ang) : -ck * __sinf(ang)) * win * (1.f / NFFT);
    }
    wb[idx] = (_Float16)v;
}

__global__ void k_zero_f32(float* __restrict__ p, long n)
{
    long i = (long)blockIdx.x * blockDim.x + threadIdx.x;
    if (i < n) p[i] = 0.f;
}

// overlap-add frames[M][1024] into acc[B][LFULL]
__global__ void k_ola(const float* __restrict__ fr, float* __restrict__ acc)
{
    long idx = (long)blockIdx.x * blockDim.x + threadIdx.x;
    if (idx >= (long)MROWS * NFFT) return;
    const long m = idx >> 10;
    const int n = (int)(idx & (NFFT - 1));
    const int b = (int)(m >> 11), t = (int)(m & (TSEQ - 1));
    atomicAdd(&acc[(long)b * LFULL + (long)t * HOP + n], fr[idx]);
}

// normalize by analytic window envelope, crop NFFT/2 each side
__global__ void k_final(const float* __restrict__ acc, float* __restrict__ out)
{
    long idx = (long)blockIdx.x * blockDim.x + threadIdx.x;
    if (idx >= (long)BATCH * LOUT) return;
    const int b = (int)(idx / LOUT);
    const long l = (idx - (long)b * LOUT) + NFFT / 2;
    int t0 = (int)((l - (NFFT - 1) + (HOP - 1)) / HOP); if (t0 < 0) t0 = 0;
    int t1 = (int)(l / HOP); if (t1 > TSEQ - 1) t1 = TSEQ - 1;
    float env = 0.f;
    for (int t = t0; t <= t1; ++t) {
        const int n = (int)(l - (long)t * HOP);
        const float w = 0.5f - 0.5f * __cosf(6.28318530717958647f * n * (1.f / NFFT));
        env += w * w;
    }
    out[idx] = acc[(long)b * LFULL + l] / ((env > 1e-11f) ? env : 1.f);
}

// =====================================================================
// host side
// =====================================================================
static inline size_t alignup(size_t x) { return (x + 255) & ~(size_t)255; }

extern "C" void kernel_launch(void* const* d_in, const int* in_sizes, int n_in,
                              void* d_out, int out_size, void* d_ws, size_t ws_size,
                              hipStream_t stream)
{
    // pytree-flatten (sorted dict keys) input mapping
    const float* act_in_la = (const float*)d_in[0];
    const float* act_in_lb = (const float*)d_in[1];
    // blocks[b] at 2+10b: act_la,act_lb,dw_b,dw_w,gamma,norm_w,pw1_b,pw1_w,pw2_b,pw2_w
    const float* conv_in_b = (const float*)d_in[42];
    const float* conv_in_w = (const float*)d_in[43];
    const float* mag_la = (const float*)d_in[44];
    const float* mag_lb = (const float*)d_in[45];
    const float* mag_l1b = (const float*)d_in[46];
    const float* mag_l1w = (const float*)d_in[47];
    const float* mag_l2b = (const float*)d_in[48];
    const float* mag_l2w = (const float*)d_in[49];
    const float* norm_w  = (const float*)d_in[50];
    const float* ph_cw   = (const float*)d_in[51];
    const float* ph_hw   = (const float*)d_in[52];
    const float* ph_bias = (const float*)d_in[53];
    const float* ph_pb   = (const float*)d_in[54];
    const float* ph_pw   = (const float*)d_in[55];
    const float* ph_sk   = (const float*)d_in[56];
    // res[0][p] at 57+6p: b1,b2,la,lb,w1,w2
    const float* skip_b = (const float*)d_in[75];
    const float* skip_w = (const float*)d_in[76];
    const float* xin    = (const float*)d_in[77];
    float* out = (float*)d_out;

    // ---- workspace arena ----
    char* ws = (char*)d_ws; size_t cur = 0;
    auto take = [&](size_t bytes) { char* p = ws + cur; cur += alignup(bytes); return p; };

    _Float16* WT_CONVIN = (_Float16*)take((size_t)7 * 256 * 256 * 2);
    _Float16* WT_SKIP   = (_Float16*)take((size_t)256 * 256 * 2);
    _Float16* WT_PW1[4], *WT_PW2[4];
    for (int b = 0; b < 4; ++b) {
        WT_PW1[b] = (_Float16*)take((size_t)1024 * 256 * 2);
        WT_PW2[b] = (_Float16*)take((size_t)256 * 1024 * 2);
    }
    _Float16* WT_R1[3], *WT_R2[3];
    for (int p = 0; p < 3; ++p) {
        WT_R1[p] = (_Float16*)take((size_t)3 * 256 * 256 * 2);
        WT_R2[p] = (_Float16*)take((size_t)3 * 256 * 256 * 2);
    }
    _Float16* WT_MAG1 = (_Float16*)take((size_t)256 * 256 * 2);
    _Float16* WT_MAG2 = (_Float16*)take((size_t)FBPAD * 256 * 2);
    _Float16* WT_PROJ = (_Float16*)take((size_t)PROJPAD * 256 * 2);
    float*    B_MAG2  = (float*)take((size_t)FBPAD * 4);
    float*    B_PROJ  = (float*)take((size_t)PROJPAD * 4);
    _Float16* BASIS   = (_Float16*)take((size_t)NFFT * KISTFT * 2);
    _Float16* P0      = (_Float16*)take((size_t)BATCH * PADT * DIMC * 2);
    _Float16* P1      = (_Float16*)take((size_t)BATCH * PADT * DIMC * 2);
    float*    HBUF    = (float*)take((size_t)MROWS * DIMC * 4);
    float*    SKIP    = (float*)take((size_t)MROWS * DIMC * 4);
    float*    T256    = (float*)take((size_t)MROWS * DIMC * 4);
    _Float16* A1024   = (_Float16*)take((size_t)MROWS * ICDIM * 2);
    char*     BIG     = (char*)take((size_t)MROWS * PROJPAD * 4);   // reused region
    float*    MAG     = (float*)take((size_t)MROWS * FBPAD * 4);
    float*    PH_G    = (float*)take((size_t)MROWS * FBINS * 4);
    float*    PH_P    = (float*)take((size_t)MROWS * FBINS * 4);
    float*    OUTACC  = (float*)take((size_t)BATCH * LFULL * 4);
    // aliases inside BIG (lifetimes disjoint)
    float*    G1024   = (float*)BIG;                        // pw1 outputs
    float*    PROJOUT = (float*)BIG;                        // phase proj output
    _Float16* SPECRI  = (_Float16*)BIG;                     // iSTFT A matrix
    float*    FRAMES  = (float*)(BIG + alignup((size_t)MROWS * KISTFT * 2));
    (void)ws_size; (void)n_in; (void)in_sizes; (void)out_size;

    const dim3 blk(256);
    auto g1 = [](long n) { return dim3((unsigned)((n + 255) / 256)); };

    // ---- weight prep (deterministic every call) ----
    k_ternarize_repack<<<1, 256, 0, stream>>>(conv_in_w, WT_CONVIN, 256, 256, 7, 256);
    k_ternarize_repack<<<1, 256, 0, stream>>>(skip_w,   WT_SKIP,   256, 256, 1, 256);
    for (int b = 0; b < 4; ++b) {
        k_ternarize_repack<<<1, 256, 0, stream>>>((const float*)d_in[9 + 10 * b],  WT_PW1[b], 1024, 256, 1, 1024);
        k_ternarize_repack<<<1, 256, 0, stream>>>((const float*)d_in[11 + 10 * b], WT_PW2[b], 256, 1024, 1, 256);
    }
    for (int p = 0; p < 3; ++p) {
        k_ternarize_repack<<<1, 256, 0, stream>>>((const float*)d_in[61 + 6 * p], WT_R1[p], 256, 256, 3, 256);
        k_ternarize_repack<<<1, 256, 0, stream>>>((const float*)d_in[62 + 6 * p], WT_R2[p], 256, 256, 3, 256);
    }
    k_ternarize_repack<<<1, 256, 0, stream>>>(mag_l1w, WT_MAG1, 256, 256, 1, 256);
    k_ternarize_repack<<<1, 256, 0, stream>>>(mag_l2w, WT_MAG2, FBINS, 256, 1, FBPAD);
    k_ternarize_repack<<<1, 256, 0, stream>>>(ph_pw,   WT_PROJ, PROJN, 256, 1, PROJPAD);
    k_padbias<<<g1(FBPAD), blk, 0, stream>>>(mag_l2b, B_MAG2, FBINS, FBPAD);
    k_padbias<<<g1(PROJPAD), blk, 0, stream>>>(ph_pb, B_PROJ, PROJN, PROJPAD);
    k_basis<<<g1((long)NFFT * KISTFT), blk, 0, stream>>>(BASIS);

    // ---- input staging ----
    k_zero_pads<<<g1(BATCH * 16 * DIMC), blk, 0, stream>>>(P0);
    k_zero_pads<<<g1(BATCH * 16 * DIMC), blk, 0, stream>>>(P1);
    k_cast_pad<<<g1((long)MROWS * DIMC), blk, 0, stream>>>(xin, P0);

    const long nMC = (long)MROWS * DIMC;
    const long nMI = (long)MROWS * ICDIM;
    auto gemm = [&](const _Float16* A, const _Float16* W, const float* bias, float* Y,
                    int Npad, int K, int tStride, int tOff, int ntaps, int dil) {
        dim3 grid(MROWS / 128, Npad / 64);
        wmma_gemm_taps<<<grid, blk, 0, stream>>>(A, W, bias, Y, Npad, K, tStride, tOff, ntaps, dil);
    };

    // skip + conv_in + snake
    gemm(P0, WT_SKIP,   skip_b,   SKIP, 256, 256, PADT, TOFF, 1, 1);
    gemm(P0, WT_CONVIN, conv_in_b, HBUF, 256, 256, PADT, TOFF, 7, 1);
    k_snake<<<g1(nMC), blk, 0, stream>>>(HBUF, act_in_la, act_in_lb, 255, nMC);

    // ---- 4 ConvNeXt blocks (dilations 1,2,4,8) ----
    const int DILS[4] = {1, 2, 4, 8};
    for (int b = 0; b < 4; ++b) {
        const float* la   = (const float*)d_in[2 + 10 * b];
        const float* lb   = (const float*)d_in[3 + 10 * b];
        const float* dwb  = (const float*)d_in[4 + 10 * b];
        const float* dww  = (const float*)d_in[5 + 10 * b];
        const float* gam  = (const float*)d_in[6 + 10 * b];
        const float* nw   = (const float*)d_in[7 + 10 * b];
        const float* p1b  = (const float*)d_in[8 + 10 * b];
        const float* p2b  = (const float*)d_in[10 + 10 * b];
        k_dwconv<<<g1(nMC), blk, 0, stream>>>(HBUF, dww, dwb, T256, DILS[b]);
        k_rmsnorm_pad<<<MROWS / 8, blk, 0, stream>>>(T256, nw, P1);
        gemm(P1, WT_PW1[b], p1b, G1024, 1024, 256, PADT, TOFF, 1, 1);
        k_snake_h<<<g1(nMI), blk, 0, stream>>>(G1024, la, lb, A1024, 1023, nMI);
        gemm(A1024, WT_PW2[b], p2b, T256, 256, 1024, TSEQ, 0, 1, 1);
        k_axpy_gamma<<<g1(nMC), blk, 0, stream>>>(HBUF, T256, gam, nMC);
    }
    k_add<<<g1(nMC), blk, 0, stream>>>(HBUF, SKIP, nMC);

    // ---- residual block (dilations 1,3,5) ----
    const int RDIL[3] = {1, 3, 5};
    for (int p = 0; p < 3; ++p) {
        const float* b1 = (const float*)d_in[57 + 6 * p];
        const float* b2 = (const float*)d_in[58 + 6 * p];
        const float* la = (const float*)d_in[59 + 6 * p];
        const float* lb = (const float*)d_in[60 + 6 * p];
        k_cast_pad<<<g1(nMC), blk, 0, stream>>>(HBUF, P1);
        gemm(P1, WT_R1[p], b1, T256, 256, 256, PADT, TOFF, 3, RDIL[p]);
        k_snake<<<g1(nMC), blk, 0, stream>>>(T256, la, lb, 255, nMC);
        k_cast_pad<<<g1(nMC), blk, 0, stream>>>(T256, P1);
        gemm(P1, WT_R2[p], b2, SKIP, 256, 256, PADT, TOFF, 3, 1);
        k_add<<<g1(nMC), blk, 0, stream>>>(HBUF, SKIP, nMC);
    }

    // ---- final norm, mag head ----
    k_rmsnorm_pad<<<MROWS / 8, blk, 0, stream>>>(HBUF, norm_w, P1);
    gemm(P1, WT_MAG1, mag_l1b, T256, 256, 256, PADT, TOFF, 1, 1);
    k_snake_h<<<g1(nMC), blk, 0, stream>>>(T256, mag_la, mag_lb, A1024, 255, nMC);
    gemm(A1024, WT_MAG2, B_MAG2, MAG, FBPAD, 256, TSEQ, 0, 1, 1);
    k_expclip<<<g1((long)MROWS * FBPAD), blk, 0, stream>>>(MAG, (long)MROWS * FBPAD);

    // ---- phase head ----
    gemm(P1, WT_PROJ, B_PROJ, PROJOUT, PROJPAD, 256, PADT, TOFF, 1, 1);
    k_headcomb<<<g1((long)MROWS * FBINS), blk, 0, stream>>>(PROJOUT, ph_hw, ph_cw, PH_G);
    k_cumsum<<<g1(BATCH * FBINS), blk, 0, stream>>>(PH_G, ph_bias, PH_P);
    k_smooth<<<g1((long)MROWS * FBINS), blk, 0, stream>>>(PH_P, ph_sk, PH_G);

    // ---- iSTFT via WMMA GEMM + overlap-add ----
    k_spec<<<g1((long)MROWS * KISTFT), blk, 0, stream>>>(MAG, PH_G, SPECRI);
    gemm(SPECRI, BASIS, nullptr, FRAMES, NFFT, KISTFT, TSEQ, 0, 1, 1);
    k_zero_f32<<<g1((long)BATCH * LFULL), blk, 0, stream>>>(OUTACC, (long)BATCH * LFULL);
    k_ola<<<g1((long)MROWS * NFFT), blk, 0, stream>>>(FRAMES, OUTACC);
    k_final<<<g1((long)BATCH * LOUT), blk, 0, stream>>>(OUTACC, out);
}
